// HGTConv_292057776514
// MI455X (gfx1250) — matrix operand order repaired and checked by reading, not compiled, not run on hardware
//
#include <hip/hip_runtime.h>

// ---------------------------------------------------------------------------
// HGTConv fused kernel for MI455X (gfx1250, wave32, WMMA bf16 16x16x32).
// One 256-thread block (8 waves) per batch element; per-batch pipeline is
// LDS-resident (156 KB pool -> 2 workgroups per 320 KB WGP).
//  - A fragments: 2x ds_load_b128 (contiguous per CDNA5 16-bit A layout),
//    or 4x global_load_b128 + cvt_pk_bf16 straight from L2 for shared weights.
//  - B operands stored transposed in LDS -> 2x ds_load_b128 per fragment.
//  - D stores into transposed buffers -> 1x ds_store_b128 per lane.
//  - All staging / zero-fill loops vectorized to b128 LDS stores.
// ---------------------------------------------------------------------------

typedef __bf16 bf16;
typedef bf16  v16bf __attribute__((ext_vector_type(16)));
typedef bf16  v8bf  __attribute__((ext_vector_type(8)));
typedef float v8f   __attribute__((ext_vector_type(8)));

#define WMMA_BF16(a, b, c) \
  __builtin_amdgcn_wmma_f32_16x16x32_bf16(false, (a), false, (b), (short)0, (c), false, false)

// ---- LDS byte offsets: one 159744-byte pool, overlaps only across disjoint
//      lifetimes (noted per stage) ----
#define KT_OFF      0        // bf16 K^T   64x128  S3 out -> S4 in
#define QT_OFF      16384    // bf16 Q^T   64x128  S3 out -> S4 in
#define MEST_OFF    32768    // bf16 msg^T 64x128  S3 out -> S5 in
#define BH_OFF      49152    // bf16 gelu(B) 128x64  S5 out -> S8a in
#define WMH_OFF     65536    // bf16 Wmsg^T 128x32   whole kernel
#define NIMT_OFF    73728    // bf16 NIM^T 5x16x128  S1 out -> S2 in
#define WWT_OFF     94208    // bf16 WW^T  5x16x64   S1 only
#define KPRE_OFF    94208    // bf16 128x96          S2 out -> S3 (reuses WWT)
#define QPRE_OFF    118784   // bf16 128x96          S2 out -> S3
#define MRE_OFF     0        // bf16 128x96          S2 out -> S3a (dies before Kt/Qt written)
#define MASKT_OFF   143360   // bf16 mask^T 64x128   S3 only (restaged at S8c)
#define SB_OFF      94208    // f32 64x64            S4 (Kpre dead)
#define AWT_OFF     110592   // bf16 64x256          S4 out -> S5 (Qpre dead)
#define RES2_OFF    143360   // bf16 32x256          S5 (maskt dead)
#define WWT2_OFF    94208    // bf16 WW^T restage    S8a
#define BM1T_OFF    104448   // bf16 Bm1^T 5x16x128  S8a out -> S8b
#define BM2H_OFF    0        // bf16 128x96          S8b out -> S8c
#define MASKT2_OFF  24576    // bf16 mask^T restage  S8c
#define SMEM_BYTES  159744

// A fragment (16x32 bf16) from row-major LDS: per-lane two contiguous 16B runs.
__device__ inline v16bf ldsA(const bf16* base, int ld, int lane) {
  const bf16* row = base + (lane & 15) * ld + ((lane >> 4) << 3);
  v8bf lo = *(const v8bf*)(row);
  v8bf hi = *(const v8bf*)(row + 16);
  return __builtin_shufflevector(lo, hi, 0,1,2,3,4,5,6,7,8,9,10,11,12,13,14,15);
}

// B fragment (32x16 bf16) from a TRANSPOSED store (Bt[n][k]): contiguous runs.
__device__ inline v16bf ldsBT(const bf16* base, int ld, int lane) {
  const bf16* row = base + (lane & 15) * ld + ((lane >> 4) << 3);
  v8bf lo = *(const v8bf*)(row);
  v8bf hi = *(const v8bf*)(row + 16);
  return __builtin_shufflevector(lo, hi, 0,1,2,3,4,5,6,7,8,9,10,11,12,13,14,15);
}

// A fragment straight from global f32 (L2-resident weights): 4x b128 + cvt.
__device__ inline v16bf gldA(const float* base, int ld, int lane) {
  const float* row = base + (lane & 15) * ld + ((lane >> 4) << 3);
  float4 x0 = *(const float4*)(row + 0);
  float4 x1 = *(const float4*)(row + 4);
  float4 y0 = *(const float4*)(row + 16);
  float4 y1 = *(const float4*)(row + 20);
  v16bf a;
  a[0]=(bf16)x0.x;  a[1]=(bf16)x0.y;  a[2]=(bf16)x0.z;  a[3]=(bf16)x0.w;
  a[4]=(bf16)x1.x;  a[5]=(bf16)x1.y;  a[6]=(bf16)x1.z;  a[7]=(bf16)x1.w;
  a[8]=(bf16)y0.x;  a[9]=(bf16)y0.y;  a[10]=(bf16)y0.z; a[11]=(bf16)y0.w;
  a[12]=(bf16)y1.x; a[13]=(bf16)y1.y; a[14]=(bf16)y1.z; a[15]=(bf16)y1.w;
  return a;
}

// D fragment -> 8 contiguous bf16 (one ds_store_b128).
__device__ inline void storeD8(bf16* dst, v8f acc) {
  v8bf p;
#pragma unroll
  for (int r = 0; r < 8; ++r) p[r] = (bf16)acc[r];
  *(v8bf*)dst = p;
}

// Vectorized zero of a 128x96 bf16 buffer (12288 elems, 6 b128 stores/thread).
__device__ inline void zfill96(bf16* buf, int tid) {
  v8bf z = {};
#pragma unroll
  for (int r = 0; r < 6; ++r) *(v8bf*)(buf + (tid + r*256)*8) = z;
}

__global__ __launch_bounds__(256) void hgt_fused(
    const float* __restrict__ node_inp, const float* __restrict__ WW,
    const float* __restrict__ maskm,    const float* __restrict__ Rg,
    const float* __restrict__ attn_mask,
    const float* __restrict__ WK, const float* __restrict__ bK,
    const float* __restrict__ WQ, const float* __restrict__ bQ,
    const float* __restrict__ WM, const float* __restrict__ bM,
    const float* __restrict__ WB, const float* __restrict__ bB,
    const float* __restrict__ Wup, const float* __restrict__ Wdn,
    const float* __restrict__ Wlf, const float* __restrict__ Wrt,
    float* __restrict__ out, float* __restrict__ aw_out)
{
  extern __shared__ char smem[];
  const int tid  = threadIdx.x;
  const int w    = tid >> 5;           // wave 0..7
  const int lane = tid & 31;
  const int bb   = blockIdx.x;         // batch element
  const int rr0  = (lane >> 4) << 3;   // C/D row base within tile
  const int cc   = lane & 15;          // C/D col within tile

  bf16* Kt     = (bf16*)(smem + KT_OFF);
  bf16* Qt     = (bf16*)(smem + QT_OFF);
  bf16* Mest   = (bf16*)(smem + MEST_OFF);
  bf16* Bh     = (bf16*)(smem + BH_OFF);
  bf16* Wmh    = (bf16*)(smem + WMH_OFF);
  bf16* NIMt   = (bf16*)(smem + NIMT_OFF);
  bf16* WWt    = (bf16*)(smem + WWT_OFF);
  bf16* Kpre   = (bf16*)(smem + KPRE_OFF);
  bf16* Qpre   = (bf16*)(smem + QPRE_OFF);
  bf16* Mre    = (bf16*)(smem + MRE_OFF);
  bf16* maskt  = (bf16*)(smem + MASKT_OFF);
  float* Sb    = (float*)(smem + SB_OFF);
  bf16* awt    = (bf16*)(smem + AWT_OFF);
  bf16* res2   = (bf16*)(smem + RES2_OFF);
  bf16* WWt2   = (bf16*)(smem + WWT2_OFF);
  bf16* Bm1t   = (bf16*)(smem + BM1T_OFF);
  bf16* Bm2h   = (bf16*)(smem + BM2H_OFF);
  bf16* maskt2 = (bf16*)(smem + MASKT2_OFF);

  // ---- Phase 0: Wmh = Wmsg^T (vectorized); WW^T staged+padded; zero Mre ----
  {
    const float* Wp[4] = {Wup, Wdn, Wlf, Wrt};
#pragma unroll
    for (int r = 0; r < 2; ++r) {          // 128x32, 8 elems (along k) / store
      int i0 = (tid + r*256) * 8;
      int q = i0 >> 5, k0 = i0 & 31;
      const float* src = Wp[q >> 5] + (q & 31) + k0*32;   // W[k][q2], stride 32
      v8bf p;
#pragma unroll
      for (int u = 0; u < 8; ++u) p[u] = (bf16)src[u*32];
      *(v8bf*)(Wmh + i0) = p;
    }
  }
  for (int i0 = tid*8; i0 < 5*16*64; i0 += 2048) {  // WWt[m][n(16)][k(64)]
    int m = i0 >> 10, n = (i0 >> 6) & 15, k0 = i0 & 63;
    v8bf p;
    if (n < 15) {
      const float* src = WW + ((bb*5 + m)*64 + k0)*15 + n;   // stride 15
#pragma unroll
      for (int u = 0; u < 8; ++u) p[u] = (bf16)src[u*15];
    } else {
#pragma unroll
      for (int u = 0; u < 8; ++u) p[u] = (bf16)0.0f;
    }
    *(v8bf*)(WWt + i0) = p;
  }
  zfill96(Mre, tid);
  __syncthreads();

  // ---- S1: NIM[m] = X (128x64) @ WW[m];  X A-frags straight from global ----
#pragma unroll
  for (int q = 0; q < 5; ++q) {
    int t = q*8 + w, m = t >> 3, mt = t & 7;
    v8f acc = {};
#pragma unroll
    for (int c = 0; c < 2; ++c)
      acc = WMMA_BF16(gldA(node_inp + mt*16*65536 + bb*64 + c*32, 65536, lane),
                      ldsBT(WWt + m*1024 + c*32, 64, lane), acc);
    storeD8(NIMt + m*2048 + cc*128 + mt*16 + rr0, acc);   // NIM^T[m][n][k]
  }
  __syncthreads();

  // ---- Phase 1.5: zero Kpre/Qpre (WWt now dead); stage mask^T ----
  zfill96(Kpre, tid);
  zfill96(Qpre, tid);
#pragma unroll
  for (int r = 0; r < 4; ++r) {            // maskt[j][k(128)], k>=75 zero
    int i0 = (tid + r*256) * 8;
    int j = i0 >> 7, k0 = i0 & 127;
    v8bf p;
#pragma unroll
    for (int u = 0; u < 8; ++u) {
      int k = k0 + u;
      p[u] = (k < 75) ? (bf16)maskm[k*64 + j] : (bf16)0.0f;
    }
    *(v8bf*)(maskt + i0) = p;
  }
  __syncthreads();

  // ---- S2: per-type projections W[m](128x128) @ NIM[m] + bias.
  //      A-frags read directly from global (L2); 120 wave-tiles, no staging. ----
  for (int q = 0; q < 15; ++q) {
    int t = q*8 + w;
    int combo = t >> 3, mt = t & 7;
    int m = combo / 3, p = combo % 3;
    const float* Wmat = (p == 0) ? WK : (p == 1) ? WQ : WM;
    const float* bvec = (p == 0) ? bK : (p == 1) ? bQ : bM;
    v8f acc = {};
#pragma unroll
    for (int c = 0; c < 4; ++c)
      acc = WMMA_BF16(gldA(Wmat + m*16384 + mt*16*128 + c*32, 128, lane),
                      ldsBT(NIMt + m*2048 + c*32, 128, lane), acc);
    if (cc < 15) {
#pragma unroll
      for (int r = 0; r < 8; ++r) {
        int row = mt*16 + rr0 + r;
        float v = acc[r] + bvec[m*128 + row];
        if (p == 0)      Kpre[row*96 + m*15 + cc] = (bf16)v;   // permute(0,2,1,3)
        else if (p == 1) Qpre[row*96 + m*15 + cc] = (bf16)v;
        else {           // faithful no-permute reshape (m,d,N) -> (128,75)
          int f = m*1920 + row*15 + cc;
          Mre[(f/75)*96 + (f%75)] = (bf16)v;
        }
      }
    }
  }
  __syncthreads();

  // ---- S3a: message^T = (Mre @ mask)^T  (Mre dies here; lives under Kt/Qt) ----
#pragma unroll
  for (int q = 0; q < 4; ++q) {
    int t = q*8 + w, mt = t >> 2, jt = t & 3;
    v8f acc = {};
#pragma unroll
    for (int c = 0; c < 3; ++c)
      acc = WMMA_BF16(ldsA(Mre + mt*16*96 + c*32, 96, lane),
                      ldsBT(maskt + jt*16*128 + c*32, 128, lane), acc);
    storeD8(Mest + (jt*16 + cc)*128 + mt*16 + rr0, acc);
  }
  __syncthreads();
  // ---- S3b: K^T, Q^T ----
#pragma unroll
  for (int q = 0; q < 8; ++q) {
    int t = q*8 + w;
    int tt = t & 31, mt = tt >> 2, jt = tt & 3;
    const bf16* Asrc = (t < 32) ? Kpre : Qpre;
    bf16* dst        = (t < 32) ? Kt : Qt;
    v8f acc = {};
#pragma unroll
    for (int c = 0; c < 3; ++c)
      acc = WMMA_BF16(ldsA(Asrc + mt*16*96 + c*32, 96, lane),
                      ldsBT(maskt + jt*16*128 + c*32, 128, lane), acc);
    storeD8(dst + (jt*16 + cc)*128 + mt*16 + rr0, acc);
  }
  __syncthreads();

  // ---- S4/S5 per head ----
  const float inv_sqrt_dk = 0.17677669529663687f;  // 1/sqrt(32)
  for (int h = 0; h < 4; ++h) {
    // scores S = R + Q^T K / sqrt(dk); and (independent) res2 = Wmsg^T @ msg_h
#pragma unroll
    for (int q = 0; q < 2; ++q) {
      int t = q*8 + w, it = t >> 2, jt = t & 3;
      v8f acc = {};
      acc = WMMA_BF16(ldsA(Qt + it*16*128 + h*32, 128, lane),
                      ldsBT(Kt + jt*16*128 + h*32, 128, lane), acc);
#pragma unroll
      for (int r = 0; r < 8; ++r) {
        int i = it*16 + rr0 + r, j = jt*16 + cc;
        Sb[i*64 + j] = Rg[((bb*4 + h)*64 + i)*64 + j] + acc[r] * inv_sqrt_dk;
      }
    }
#pragma unroll
    for (int q = 0; q < 4; ++q) {  // res2 flat (32x256) == res (128x64)
      int t = q*8 + w, qt = t >> 2, nt = t & 3;
      v8f acc = {};
      acc = WMMA_BF16(ldsA(Wmh + qt*16*32, 32, lane),
                      ldsBT(Mest + nt*16*128 + h*32, 128, lane), acc);
#pragma unroll
      for (int r = 0; r < 8; ++r)
        res2[(qt*16 + rr0 + r)*64 + nt*16 + cc] = (bf16)acc[r];
    }
    __syncthreads();
    if (tid < 64) {  // row softmax + NaN scrub (float4 LDS, hw exp)
      float4* row4 = (float4*)(Sb + tid*64);
      float mx = -3.402823466e+38f;
#pragma unroll
      for (int j = 0; j < 16; ++j) {
        float4 v = row4[j];
        mx = fmaxf(fmaxf(fmaxf(mx, v.x), fmaxf(v.y, v.z)), v.w);
      }
      float s = 0.0f;
#pragma unroll
      for (int j = 0; j < 16; ++j) {
        float4 v = row4[j];
        v.x = __expf(v.x - mx); v.y = __expf(v.y - mx);
        v.z = __expf(v.z - mx); v.w = __expf(v.w - mx);
        s += (v.x + v.y) + (v.z + v.w);
        row4[j] = v;
      }
      float inv = 1.0f / s;
#pragma unroll
      for (int j = 0; j < 16; ++j) {
        float4 v = row4[j];
        v.x *= inv; v.y *= inv; v.z *= inv; v.w *= inv;
        v.x = (v.x == v.x) ? v.x : 0.0f;  v.y = (v.y == v.y) ? v.y : 0.0f;
        v.z = (v.z == v.z) ? v.z : 0.0f;  v.w = (v.w == v.w) ? v.w : 0.0f;
        row4[j] = v;
      }
    }
    __syncthreads();
    // awt[i][t] = softmax[i][t%64] * attn_mask[i][t]  (vectorized)
#pragma unroll
    for (int rep = 0; rep < 8; ++rep) {
      int i0 = rep*2048 + tid*8;
      int ii = i0 >> 8, t0 = i0 & 255;
      const float4 s0 = *(const float4*)(Sb + ii*64 + (t0 & 63));
      const float4 s1 = *(const float4*)(Sb + ii*64 + (t0 & 63) + 4);
      const float4 m0 = *(const float4*)(attn_mask + i0);
      const float4 m1 = *(const float4*)(attn_mask + i0 + 4);
      float v0 = s0.x*m0.x, v1 = s0.y*m0.y, v2 = s0.z*m0.z, v3 = s0.w*m0.w;
      float v4 = s1.x*m1.x, v5 = s1.y*m1.y, v6 = s1.z*m1.z, v7 = s1.w*m1.w;
      v8bf p;
      p[0]=(bf16)v0; p[1]=(bf16)v1; p[2]=(bf16)v2; p[3]=(bf16)v3;
      p[4]=(bf16)v4; p[5]=(bf16)v5; p[6]=(bf16)v6; p[7]=(bf16)v7;
      *(v8bf*)(awt + i0) = p;
      if (bb == 0 && h == 0) {                       // second output: aw[0,0]
        *(float4*)(aw_out + i0)     = make_float4(v0, v1, v2, v3);
        *(float4*)(aw_out + i0 + 4) = make_float4(v4, v5, v6, v7);
      }
    }
    __syncthreads();
    // B_h[k][i] = gelu( sum_t res2[k][t] * awt[i][t] ), K = 256
    {
      int kt = w >> 2, it = w & 3;
      v8f acc = {};
#pragma unroll
      for (int c = 0; c < 8; ++c)
        acc = WMMA_BF16(ldsA(res2 + kt*16*256 + c*32, 256, lane),
                        ldsBT(awt + it*16*256 + c*32, 256, lane), acc);
#pragma unroll
      for (int r = 0; r < 8; ++r) {
        float x = acc[r];
        float g = 0.5f * x * (1.0f + erff(x * 0.70710678118654752f));
        Bh[(h*32 + kt*16 + rr0 + r)*64 + it*16 + cc] = (bf16)g;
      }
    }
    __syncthreads();
  }

  // ---- S8a: Bm1^T[m] = (gelu(B) @ WW[m])^T   (WW^T restaged; L2-resident) ----
  for (int i0 = tid*8; i0 < 5*16*64; i0 += 2048) {
    int m = i0 >> 10, n = (i0 >> 6) & 15, k0 = i0 & 63;
    v8bf p;
    if (n < 15) {
      const float* src = WW + ((bb*5 + m)*64 + k0)*15 + n;
#pragma unroll
      for (int u = 0; u < 8; ++u) p[u] = (bf16)src[u*15];
    } else {
#pragma unroll
      for (int u = 0; u < 8; ++u) p[u] = (bf16)0.0f;
    }
    *(v8bf*)(WWt2 + i0) = p;
  }
  zfill96(Bm2h, tid);   // Kt region dead
  __syncthreads();
#pragma unroll
  for (int q = 0; q < 5; ++q) {
    int t = q*8 + w, m = t >> 3, mt = t & 7;
    v8f acc = {};
#pragma unroll
    for (int c = 0; c < 2; ++c)
      acc = WMMA_BF16(ldsA(Bh + mt*16*64 + c*32, 64, lane),
                      ldsBT(WWt2 + m*1024 + c*32, 64, lane), acc);
    storeD8(Bm1t + m*2048 + cc*128 + mt*16 + rr0, acc);
  }
  __syncthreads();

  // ---- S8b: Bm2 = WB[m] @ Bm1[m] + bB (A-frags from global), permuted scatter;
  //      mask^T restaged in parallel (regions disjoint) ----
#pragma unroll
  for (int q = 0; q < 5; ++q) {
    int t = q*8 + w, m = t >> 3, mt = t & 7;
    v8f acc = {};
#pragma unroll
    for (int c = 0; c < 4; ++c)
      acc = WMMA_BF16(gldA(WB + m*16384 + mt*16*128 + c*32, 128, lane),
                      ldsBT(Bm1t + m*2048 + c*32, 128, lane), acc);
    if (cc < 15) {
#pragma unroll
      for (int r = 0; r < 8; ++r) {
        int row = mt*16 + rr0 + r;
        Bm2h[row*96 + m*15 + cc] = (bf16)(acc[r] + bB[m*128 + row]);
      }
    }
  }
#pragma unroll
  for (int r = 0; r < 4; ++r) {   // maskt2 (Qt/Mest regions dead)
    int i0 = (tid + r*256) * 8;
    int j = i0 >> 7, k0 = i0 & 127;
    v8bf p;
#pragma unroll
    for (int u = 0; u < 8; ++u) {
      int k = k0 + u;
      p[u] = (k < 75) ? (bf16)maskm[k*64 + j] : (bf16)0.0f;
    }
    *(v8bf*)(maskt2 + i0) = p;
  }
  __syncthreads();

  // ---- S8c: out = Bm2 @ mask + node_inp ----
#pragma unroll
  for (int q = 0; q < 4; ++q) {
    int t = q*8 + w, mt = t >> 2, jt = t & 3;
    v8f acc = {};
#pragma unroll
    for (int c = 0; c < 3; ++c)
      acc = WMMA_BF16(ldsA(Bm2h + mt*16*96 + c*32, 96, lane),
                      ldsBT(maskt2 + jt*16*128 + c*32, 128, lane), acc);
#pragma unroll
    for (int r = 0; r < 8; ++r) {
      int d = mt*16 + rr0 + r, j = jt*16 + cc;
      int idx = d*65536 + bb*64 + j;           // out is (d, b, n)
      out[idx] = acc[r] + node_inp[idx];
    }
  }
}

extern "C" void kernel_launch(void* const* d_in, const int* in_sizes, int n_in,
                              void* d_out, int out_size, void* d_ws, size_t ws_size,
                              hipStream_t stream) {
  (void)in_sizes; (void)n_in; (void)d_ws; (void)ws_size; (void)out_size;
  const float* node_inp  = (const float*)d_in[0];
  const float* WW        = (const float*)d_in[1];
  const float* maskm     = (const float*)d_in[2];
  const float* Rg        = (const float*)d_in[3];
  const float* attn_mask = (const float*)d_in[4];
  const float* WK  = (const float*)d_in[5];
  const float* bK  = (const float*)d_in[6];
  const float* WQ  = (const float*)d_in[7];
  const float* bQ  = (const float*)d_in[8];
  const float* WM  = (const float*)d_in[9];
  const float* bM  = (const float*)d_in[10];
  const float* WB  = (const float*)d_in[11];
  const float* bB  = (const float*)d_in[12];
  const float* Wup = (const float*)d_in[13];
  const float* Wdn = (const float*)d_in[14];
  const float* Wlf = (const float*)d_in[15];
  const float* Wrt = (const float*)d_in[16];

  float* out    = (float*)d_out;
  float* aw_out = out + 128*1024*64;   // second tuple output: aw[0,0] (64x256)

  // 156 KB dynamic LDS (2 workgroups per 320 KB WGP).
  (void)hipFuncSetAttribute((const void*)hgt_fused,
                            hipFuncAttributeMaxDynamicSharedMemorySize, SMEM_BYTES);

  hipLaunchKernelGGL(hgt_fused, dim3(1024), dim3(256), SMEM_BYTES, stream,
                     node_inp, WW, maskm, Rg, attn_mask,
                     WK, bK, WQ, bQ, WM, bM, WB, bB,
                     Wup, Wdn, Wlf, Wrt, out, aw_out);
}
